// CNN3_P_3719441679180
// MI455X (gfx1250) — matrix operand order repaired and checked by reading, not compiled, not run on hardware
//
#include <hip/hip_runtime.h>
#include <hip/hip_bf16.h>

// ---------------------------------------------------------------------------
// Problem constants (from reference)
// ---------------------------------------------------------------------------
#define BATCH   2048
#define CL      128
#define IL      64
#define PC      256
#define CH      256     // CH1 == CH2 == CH3 == 256
#define KCONV   768     // 256 channels * 3 taps
#define F1      512
#define LPSEUDO 127     // pseudo-conv output length
#define L1      125
#define L2      123
#define L3      121
#define F1IN    (CH * L3)   // 30976
#define HROWS   132         // 128 padded positions + 4 halo/zero rows

#define KC      64          // FC1 K-chunk staged in LDS
#define FC1_NB  256         // FC1 N columns per block

typedef __attribute__((ext_vector_type(16))) _Float16 v16h;
typedef __attribute__((ext_vector_type(8)))  _Float16 v8h;
typedef __attribute__((ext_vector_type(8)))  float    v8f;
typedef __attribute__((ext_vector_type(4)))  unsigned int v4u;
typedef __attribute__((ext_vector_type(4)))  int      v4i;
typedef __attribute__((ext_vector_type(8)))  int      v8i;

#if __has_builtin(__builtin_amdgcn_tensor_load_to_lds) && \
    __has_builtin(__builtin_amdgcn_s_wait_tensorcnt)
#define HAVE_TDM 1
#else
#define HAVE_TDM 0
#endif

// ---------------------------------------------------------------------------
// WMMA helpers
// ---------------------------------------------------------------------------
__device__ __forceinline__ v8f wmma_f16(v16h a, v16h b, v8f c) {
  // D = A(16x32 f16) * B(32x16 f16) + C(16x16 f32)
  return __builtin_amdgcn_wmma_f32_16x16x32_f16(
      /*neg_a=*/false, a, /*neg_b=*/false, b,
      /*c_mod=*/(short)0, c, /*reuse_a=*/false, /*reuse_b=*/false);
}

__device__ __forceinline__ v8f vzero8() {
  v8f z;
#pragma unroll
  for (int i = 0; i < 8; ++i) z[i] = 0.0f;
  return z;
}

// A fragment: matrix stored row-major [rows][ld] (f16), tile at (row0, k0).
// ISA 16-bit A layout: lanes 0-15 hold K {0..7,16..23}, lanes 16-31 hold
// K {8..15,24..31} of row (lane&15).
__device__ __forceinline__ v16h load_fragA(const _Float16* __restrict__ base,
                                           int ld, int row0, int k0, int lane) {
  int r    = row0 + (lane & 15);
  int koff = (lane >> 4) * 8;
  const _Float16* p = base + (size_t)r * ld + k0 + koff;
  v8h lo = *(const v8h*)(p);
  v8h hi = *(const v8h*)(p + 16);
  return __builtin_shufflevector(lo, hi, 0,1,2,3,4,5,6,7,8,9,10,11,12,13,14,15);
}

// B fragment from a transposed panel Bt[n][k] (f16).
__device__ __forceinline__ v16h load_fragB(const _Float16* __restrict__ base,
                                           int ld, int n0, int k0, int lane) {
  int r    = n0 + (lane & 15);
  int koff = (lane >> 4) * 8;
  const _Float16* p = base + (size_t)r * ld + k0 + koff;
  v8h lo = *(const v8h*)(p);
  v8h hi = *(const v8h*)(p + 16);
  return __builtin_shufflevector(lo, hi, 0,1,2,3,4,5,6,7,8,9,10,11,12,13,14,15);
}

// ---------------------------------------------------------------------------
// TDM: one 2D tile (tile_d1 rows x tile_d0 f16 elements) -> LDS, via D#
// descriptor groups per CDNA5 ISA §8.3/8.4. data_size=2B, type=2 ("image").
// This toolchain exposes the 6-argument builtin:
//   (v4u g0, v8i g1, v4i g2, v4i g3, v8i g4, i32 cpol)
// ---------------------------------------------------------------------------
#if HAVE_TDM
__device__ __forceinline__ void tdm_load_2d_f16(
    unsigned lds_byte_off, const void* gptr,
    unsigned tensor_d0, unsigned tensor_d1,
    unsigned tile_d0, unsigned tile_d1,
    unsigned long long row_stride_elems)
{
  unsigned long long ga = (unsigned long long)(size_t)gptr;
  v4u g0;
  g0[0] = 1u;                                        // count=1, user mode
  g0[1] = lds_byte_off;                              // lds_addr
  g0[2] = (unsigned)(ga & 0xFFFFFFFFull);            // global_addr[31:0]
  g0[3] = (unsigned)((ga >> 32) & 0x01FFFFFFull)     // global_addr[56:32]
        | 0x80000000u;                               // type=2
  v8i g1;
  g1[0] = (int)(1u << 16);                           // data_size=1 (2 bytes)
  g1[1] = (int)((tensor_d0 & 0xFFFFu) << 16);        // tensor_dim0[15:0]
  g1[2] = (int)((tensor_d0 >> 16) & 0xFFFFu)
        | (int)((tensor_d1 & 0xFFFFu) << 16);        // dim0 hi, dim1 lo
  g1[3] = (int)((tensor_d1 >> 16) & 0xFFFFu)
        | (int)((tile_d0 & 0xFFFFu) << 16);          // dim1 hi, tile_dim0
  g1[4] = (int)(tile_d1 & 0xFFFFu);                  // tile_dim1; tile_dim2=0
  g1[5] = (int)(row_stride_elems & 0xFFFFFFFFull);   // dim0_stride[31:0]
  g1[6] = (int)((row_stride_elems >> 32) & 0xFFFFull); // dim0_stride[47:32]
  g1[7] = 0;                                         // dim1_stride unused (2D)
  v4i g2 = {0, 0, 0, 0};                             // 2D: groups 2/3 unused
  v4i g3 = {0, 0, 0, 0};
  v8i g4 = {0, 0, 0, 0, 0, 0, 0, 0};                 // extra group (clang-23)
  __builtin_amdgcn_tensor_load_to_lds(g0, g1, g2, g3, g4, 0);
}
#endif

// ---------------------------------------------------------------------------
// Weight repack kernels (f32 -> f16 GEMM layouts)
// ---------------------------------------------------------------------------
__global__ void k_pack_wp(const float* __restrict__ w, _Float16* __restrict__ o) {
  int i = blockIdx.x * blockDim.x + threadIdx.x;
  if (i >= PC * 128) return;
  int m = i >> 7, q = i & 127;
  float v = (q < 64) ? w[m * 128 + q * 2] : w[m * 128 + (q - 64) * 2 + 1];
  o[i] = (_Float16)v;
}

__global__ void k_pack_conv(const float* __restrict__ w, _Float16* __restrict__ o) {
  int i = blockIdx.x * blockDim.x + threadIdx.x;
  if (i >= CH * KCONV) return;
  int m = i / KCONV, rem = i - m * KCONV;
  int t = rem >> 8, ci = rem & 255;
  o[i] = (_Float16)w[m * KCONV + ci * 3 + t];
}

__global__ void k_f32_to_f16(const float* __restrict__ s, _Float16* __restrict__ d, int n) {
  int i = blockIdx.x * blockDim.x + threadIdx.x;
  if (i < n) d[i] = (_Float16)s[i];
}

// ---------------------------------------------------------------------------
// Fused front: pseudo-conv + 3 conv layers, one sample per workgroup.
// ---------------------------------------------------------------------------
__global__ __launch_bounds__(256) void fused_front(
    const float*    __restrict__ x,
    const _Float16* __restrict__ Ap,
    const float*    __restrict__ bp,
    const _Float16* __restrict__ A1, const float* __restrict__ b1,
    const _Float16* __restrict__ A2, const float* __restrict__ b2,
    const _Float16* __restrict__ A3, const float* __restrict__ b3,
    _Float16*       __restrict__ h3g)
{
  __shared__ _Float16 hbuf[HROWS * CH];   // activations [pos][chan]
  __shared__ _Float16 xbuf[128 * 128];    // pseudo-layer B^T panel [n][k]

  const int b    = blockIdx.x;
  const int tid  = threadIdx.x;
  const int lane = tid & 31;
  const int wave = tid >> 5;
  const int m0   = wave * 32;
  const float* xs = x + (size_t)b * (CL * IL);

  // ---- Phase 0: build x panel; zero halo rows of hbuf --------------------
  {
    int n    = tid >> 1;
    int half = (tid & 1) * 64;
#pragma unroll 8
    for (int j = 0; j < 64; ++j) {
      float v = 0.0f;
      if (n < LPSEUDO) v = (half == 0) ? xs[j] : xs[(n + 1) * IL + j];
      xbuf[n * 128 + half + j] = (_Float16)v;
    }
#pragma unroll 4
    for (int j = 0; j < 4; ++j)
      hbuf[128 * CH + tid * 4 + j] = (_Float16)0.0f;
  }
  __syncthreads();

  v8f acc[2][8];

  // ---- Phase 1: pseudo-conv GEMM (M=256, N=128, K=128) -------------------
#pragma unroll
  for (int mt = 0; mt < 2; ++mt)
#pragma unroll
    for (int nt = 0; nt < 8; ++nt) acc[mt][nt] = vzero8();

#pragma unroll
  for (int k0 = 0; k0 < 128; k0 += 32) {
    v16h a0 = load_fragA(Ap, 128, m0,      k0, lane);
    v16h a1 = load_fragA(Ap, 128, m0 + 16, k0, lane);
#pragma unroll
    for (int nt = 0; nt < 8; ++nt) {
      v16h bf = load_fragB(xbuf, 128, nt * 16, k0, lane);
      acc[0][nt] = wmma_f16(a0, bf, acc[0][nt]);
      acc[1][nt] = wmma_f16(a1, bf, acc[1][nt]);
    }
  }
  __syncthreads();

  {
    const int Lvalid = LPSEUDO;
#pragma unroll
    for (int mt = 0; mt < 2; ++mt)
#pragma unroll
      for (int nt = 0; nt < 8; ++nt)
#pragma unroll
        for (int r = 0; r < 8; ++r) {
          int m = m0 + mt * 16 + r + ((lane >> 4) * 8);
          int n = nt * 16 + (lane & 15);
          float v = 0.0f;
          if (n < Lvalid) {
            v = acc[mt][nt][r] + bp[m];
            v = v > 0.0f ? v : 0.0f;
          }
          hbuf[n * CH + m] = (_Float16)v;
        }
  }
  __syncthreads();

  // ---- Phases 2..4: conv layers (M=256, N=128, K=768) --------------------
  const _Float16* As[3]   = {A1, A2, A3};
  const float*    bs[3]   = {b1, b2, b3};
  const int       Lout[3] = {L1, L2, L3};

  for (int layer = 0; layer < 3; ++layer) {
    const _Float16* A = As[layer];

#pragma unroll
    for (int mt = 0; mt < 2; ++mt)
#pragma unroll
      for (int nt = 0; nt < 8; ++nt) acc[mt][nt] = vzero8();

    for (int k0 = 0; k0 < KCONV; k0 += 32) {
      int t   = k0 >> 8;
      int ci0 = k0 & 255;
      v16h a0 = load_fragA(A, KCONV, m0,      k0, lane);
      v16h a1 = load_fragA(A, KCONV, m0 + 16, k0, lane);
      const _Float16* bbase = hbuf + t * CH;
#pragma unroll
      for (int nt = 0; nt < 8; ++nt) {
        v16h bf = load_fragB(bbase, CH, nt * 16, ci0, lane);
        acc[0][nt] = wmma_f16(a0, bf, acc[0][nt]);
        acc[1][nt] = wmma_f16(a1, bf, acc[1][nt]);
      }
    }
    __syncthreads();

    if (layer < 2) {
      const float* bias = bs[layer];
      const int Lvalid  = Lout[layer];
#pragma unroll
      for (int mt = 0; mt < 2; ++mt)
#pragma unroll
        for (int nt = 0; nt < 8; ++nt)
#pragma unroll
          for (int r = 0; r < 8; ++r) {
            int m = m0 + mt * 16 + r + ((lane >> 4) * 8);
            int n = nt * 16 + (lane & 15);
            float v = 0.0f;
            if (n < Lvalid) {
              v = acc[mt][nt][r] + bias[m];
              v = v > 0.0f ? v : 0.0f;
            }
            hbuf[n * CH + m] = (_Float16)v;
          }
      __syncthreads();
    } else {
      _Float16* out = h3g + (size_t)b * F1IN;
#pragma unroll
      for (int mt = 0; mt < 2; ++mt)
#pragma unroll
        for (int nt = 0; nt < 8; ++nt)
#pragma unroll
          for (int r = 0; r < 8; ++r) {
            int m = m0 + mt * 16 + r + ((lane >> 4) * 8);
            int n = nt * 16 + (lane & 15);
            if (n < L3) {
              float v = acc[mt][nt][r] + b3[m];
              v = v > 0.0f ? v : 0.0f;
              out[m * L3 + n] = (_Float16)v;
            }
          }
    }
  }
}

// ---------------------------------------------------------------------------
// FC1 GEMM: out(2048x512) = relu(H(2048x30976) * Wf1^T + bf1)
// Block tile 64(M) x 256(N); Wf1 K-chunks staged in LDS (TDM double buffer).
// H3 (127MB, > L2) is streamed exactly F1/FC1_NB = 2 times from HBM.
// ---------------------------------------------------------------------------
__device__ __forceinline__ void fc1_fetch_chunk(const _Float16* __restrict__ Wn,
                                                _Float16* Bs_dst, int nblk,
                                                int c, int tid, int wave) {
  const _Float16* gsrc = Wn + (size_t)nblk * F1IN + (size_t)c * KC;
#if HAVE_TDM
  if (wave == 0) {
    tdm_load_2d_f16((unsigned)(size_t)Bs_dst, gsrc,
                    /*tensor_d0=*/F1IN, /*tensor_d1=*/F1,
                    /*tile_d0=*/KC, /*tile_d1=*/FC1_NB,
                    /*row_stride=*/(unsigned long long)F1IN);
  }
#else
  (void)wave;
  const _Float16* s = gsrc + (size_t)tid * F1IN;   // thread t copies row t
  _Float16*       d = Bs_dst + tid * KC;
#pragma unroll
  for (int j = 0; j < KC; j += 8)
    *(v8h*)(d + j) = *(const v8h*)(s + j);
#endif
}

__device__ __forceinline__ void fc1_wait_chunk(int wave) {
#if HAVE_TDM
  if (wave == 0) __builtin_amdgcn_s_wait_tensorcnt((short)0);
#else
  (void)wave;
#endif
}

__global__ __launch_bounds__(256) void fc1_gemm(
    const _Float16* __restrict__ H,     // [2048][30976]
    const _Float16* __restrict__ Wn,    // [512][30976]
    const float*    __restrict__ bias,  // [512]
    float*          __restrict__ out)   // [2048][512]
{
  __shared__ _Float16 Bs[2][FC1_NB * KC];   // 2 x 32KB double buffer

  const int tid  = threadIdx.x;
  const int lane = tid & 31;
  const int wave = tid >> 5;
  const int mblk = blockIdx.x * 64;
  const int nblk = blockIdx.y * FC1_NB;
  const int m0   = mblk + (wave >> 2) * 32;   // 2 wave rows in M
  const int n0w  = (wave & 3) * 64;           // 4 wave cols in N (64 each)

  v8f acc[2][4];
#pragma unroll
  for (int i = 0; i < 2; ++i)
#pragma unroll
    for (int j = 0; j < 4; ++j) acc[i][j] = vzero8();

  const int NCHUNK = F1IN / KC;   // 484

  fc1_fetch_chunk(Wn, &Bs[0][0], nblk, 0, tid, wave);

  for (int c = 0; c < NCHUNK; ++c) {
    const int buf = c & 1;
    fc1_wait_chunk(wave);
    __syncthreads();                          // chunk c visible to all
    if (c + 1 < NCHUNK)
      fc1_fetch_chunk(Wn, &Bs[buf ^ 1][0], nblk, c + 1, tid, wave);

    const int kbase = c * KC;
#pragma unroll
    for (int kk = 0; kk < KC; kk += 32) {
      __builtin_prefetch(H + (size_t)m0 * F1IN + kbase + kk + 1024, 0, 1);
      v16h a0 = load_fragA(H, F1IN, m0,      kbase + kk, lane);
      v16h a1 = load_fragA(H, F1IN, m0 + 16, kbase + kk, lane);
#pragma unroll
      for (int nt = 0; nt < 4; ++nt) {
        v16h bf = load_fragB(&Bs[buf][0], KC, n0w + nt * 16, kk, lane);
        acc[0][nt] = wmma_f16(a0, bf, acc[0][nt]);
        acc[1][nt] = wmma_f16(a1, bf, acc[1][nt]);
      }
    }
    __syncthreads();                          // done reading buf
  }

#pragma unroll
  for (int mt = 0; mt < 2; ++mt)
#pragma unroll
    for (int nt = 0; nt < 4; ++nt)
#pragma unroll
      for (int r = 0; r < 8; ++r) {
        int m = m0 + mt * 16 + r + ((lane >> 4) * 8);
        int n = nblk + n0w + nt * 16 + (lane & 15);
        float v = acc[mt][nt][r] + bias[n];
        v = v > 0.0f ? v : 0.0f;
        out[(size_t)m * F1 + n] = v;
      }
}

// ---------------------------------------------------------------------------
// FC2: out[b] = fc1[b,:] . Wf2 + bf2   (one wave per row)
// ---------------------------------------------------------------------------
__global__ __launch_bounds__(256) void fc2_gemv(
    const float* __restrict__ fc1,
    const float* __restrict__ w,
    const float* __restrict__ bias,
    float*       __restrict__ out)
{
  int row  = blockIdx.x * 8 + (threadIdx.x >> 5);
  int lane = threadIdx.x & 31;
  const float* r = fc1 + (size_t)row * F1;
  float s = 0.0f;
#pragma unroll
  for (int i = lane; i < F1; i += 32) s += r[i] * w[i];
#pragma unroll
  for (int off = 16; off > 0; off >>= 1) s += __shfl_down(s, off, 32);
  if (lane == 0) out[row] = s + bias[0];
}

// ---------------------------------------------------------------------------
// Workspace layout (bytes), all 16B-aligned
// ---------------------------------------------------------------------------
static constexpr size_t OFF_AP   = 0;
static constexpr size_t OFF_A1   = OFF_AP   + (size_t)PC * 128 * 2;
static constexpr size_t OFF_A2   = OFF_A1   + (size_t)CH * KCONV * 2;
static constexpr size_t OFF_A3   = OFF_A2   + (size_t)CH * KCONV * 2;
static constexpr size_t OFF_WF1H = OFF_A3   + (size_t)CH * KCONV * 2;
static constexpr size_t OFF_H3   = OFF_WF1H + (size_t)F1 * F1IN * 2;
static constexpr size_t OFF_FC1  = OFF_H3   + (size_t)BATCH * F1IN * 2;

extern "C" void kernel_launch(void* const* d_in, const int* in_sizes, int n_in,
                              void* d_out, int out_size, void* d_ws, size_t ws_size,
                              hipStream_t stream) {
  const float* x   = (const float*)d_in[0];
  const float* Wp  = (const float*)d_in[1];
  const float* bp  = (const float*)d_in[2];
  const float* W1  = (const float*)d_in[3];
  const float* b1  = (const float*)d_in[4];
  const float* W2  = (const float*)d_in[5];
  const float* b2  = (const float*)d_in[6];
  const float* W3  = (const float*)d_in[7];
  const float* b3  = (const float*)d_in[8];
  const float* Wf1 = (const float*)d_in[9];
  const float* bf1 = (const float*)d_in[10];
  const float* Wf2 = (const float*)d_in[11];
  const float* bf2 = (const float*)d_in[12];
  float*       out = (float*)d_out;

  char* ws = (char*)d_ws;
  _Float16* Ap   = (_Float16*)(ws + OFF_AP);
  _Float16* A1   = (_Float16*)(ws + OFF_A1);
  _Float16* A2   = (_Float16*)(ws + OFF_A2);
  _Float16* A3   = (_Float16*)(ws + OFF_A3);
  _Float16* Wf1h = (_Float16*)(ws + OFF_WF1H);
  _Float16* H3   = (_Float16*)(ws + OFF_H3);
  float*    FC1o = (float*)(ws + OFF_FC1);

  k_pack_wp  <<<(PC * 128   + 255) / 256, 256, 0, stream>>>(Wp, Ap);
  k_pack_conv<<<(CH * KCONV + 255) / 256, 256, 0, stream>>>(W1, A1);
  k_pack_conv<<<(CH * KCONV + 255) / 256, 256, 0, stream>>>(W2, A2);
  k_pack_conv<<<(CH * KCONV + 255) / 256, 256, 0, stream>>>(W3, A3);
  {
    int n = F1 * F1IN;
    k_f32_to_f16<<<(n + 255) / 256, 256, 0, stream>>>(Wf1, Wf1h, n);
  }

  fused_front<<<BATCH, 256, 0, stream>>>(x, Ap, bp, A1, b1, A2, b2, A3, b3, H3);

  fc1_gemm<<<dim3(BATCH / 64, F1 / FC1_NB), 256, 0, stream>>>(H3, Wf1h, bf1, FC1o);

  fc2_gemv<<<BATCH / 8, 256, 0, stream>>>(FC1o, Wf2, bf2, out);
}
